// LlamaSdpaAttention_5806795784785
// MI455X (gfx1250) — matrix operand order repaired and checked by reading, not compiled, not run on hardware
//
#include <hip/hip_runtime.h>
#include <stdint.h>

// ---------------- problem constants ----------------
#define B_   2
#define S_   2048
#define H_   2048
#define NH_  32
#define NKV_ 8
#define HD_  64
#define RANK_ 16
#define DQ_  (NH_ * HD_)   // 2048
#define DKV_ (NKV_ * HD_)  // 512
#define M_   (B_ * S_)     // 4096
#define LORA_SCALE (1.0f / 16.0f)

typedef _Float16 half8  __attribute__((ext_vector_type(8)));
typedef _Float16 half16 __attribute__((ext_vector_type(16)));
typedef float    float8 __attribute__((ext_vector_type(8)));
typedef int      i32x4  __attribute__((ext_vector_type(4)));

// pointer-to-int4 in global (AS1) / LDS (AS3) address spaces, for the async builtin
typedef __attribute__((address_space(1))) i32x4* as1_i4p;
typedef __attribute__((address_space(3))) i32x4* as3_i4p;

#if defined(__has_builtin)
#  if __has_builtin(__builtin_amdgcn_global_load_async_to_lds_b128)
#    define HAVE_ASYNC_LDS 1
#  endif
#endif
#ifndef HAVE_ASYNC_LDS
#  define HAVE_ASYNC_LDS 0
#endif

// ---------------- WMMA fragment loaders (CDNA5 f16 16x16x32 layouts) ----------------
// A (16x32, MxK): lane L<16 -> row L, K kb..kb+15;
//                 lane L>=16 -> row L-16, K kb+16..kb+31
__device__ __forceinline__ half16 load_afrag(const _Float16* p, int ld, int row0, int kb, int lane) {
  const int r = lane & 15, hl = lane >> 4;
  const _Float16* q = p + (size_t)(row0 + r) * ld + kb + hl * 16;
  const half8 lo = *(const half8*)q;
  const half8 hi = *(const half8*)(q + 8);
  half16 out;
#pragma unroll
  for (int i = 0; i < 8; ++i) { out[i] = lo[i]; out[i + 8] = hi[i]; }
  return out;
}

// B (32x16, KxN) from row-major Bt[N][K]: lane r / r+16 hold column n0+r;
// lanes 0-15 carry K kb..kb+15, lanes 16-31 carry K kb+16..kb+31 (contiguous 32B/lane).
__device__ __forceinline__ half16 load_bfrag(const _Float16* p, int ld, int n0, int kb, int lane) {
  const int r = lane & 15, hl = lane >> 4;
  return *(const half16*)(p + (size_t)(n0 + r) * ld + kb + hl * 16);
}

__device__ __forceinline__ float8 wmma_f16(half16 a, half16 b, float8 c) {
  return __builtin_amdgcn_wmma_f32_16x16x32_f16(false, a, false, b, (short)0, c, false, false);
}

// ---------------- LoRA fold: W_eff[o][i] = W[o][i] + SCALE * (A@B)[i][o], f16 out ----------------
__global__ void prep_weight(const float* __restrict__ W, const float* __restrict__ A,
                            const float* __restrict__ Bm, _Float16* __restrict__ Wt,
                            int din, int dout) {
  const size_t idx = (size_t)blockIdx.x * blockDim.x + threadIdx.x;
  if (idx >= (size_t)din * dout) return;
  const int o = (int)(idx / din);
  const int i = (int)(idx % din);
  float acc = 0.0f;
#pragma unroll
  for (int r = 0; r < RANK_; ++r) acc += A[(size_t)i * RANK_ + r] * Bm[(size_t)r * dout + o];
  Wt[idx] = (_Float16)(W[idx] + LORA_SCALE * acc);
}

__global__ void to_half_kernel(const float* __restrict__ src, _Float16* __restrict__ dst, int n) {
  const int idx = blockIdx.x * blockDim.x + threadIdx.x;
  if (idx < n) dst[idx] = (_Float16)src[idx];
}

// ---------------- block-cooperative WMMA GEMM ----------------
// C[M,N] f32 = Ah[M,K] @ Bt[N,K]^T.  Block = 8 waves = 256x64 C tile.
// The 64-col B strip (64x32 f16 per K-step) is staged once per block into LDS
// (async global->LDS when available), shared by all 8 waves; each wave does a
// 32x64 register tile: 2 A-frags x 4 B-frags -> 8 WMMAs per K-step.
#define BROWPAD 40  // 32 + 8-half pad: 80B row stride keeps 16B alignment, kills bank conflicts
__global__ __launch_bounds__(256) void gemm_wmma(const _Float16* __restrict__ A,
                                                 const _Float16* __restrict__ Bt,
                                                 float* __restrict__ C,
                                                 int Mdim, int Ndim, int Kdim) {
  __shared__ __align__(128) _Float16 btile[64 * BROWPAD];
  const int tid  = threadIdx.x;
  const int lane = tid & 31;
  const int wib  = tid >> 5;            // wave in block: 0..7
  const int ntiles = Ndim >> 6;
  const int bm = blockIdx.x / ntiles;
  const int bn = blockIdx.x % ntiles;
  const int m0 = bm * 256 + wib * 32;
  const int n0 = bn * 64;
  if (m0 >= Mdim) return;
  const int r = lane & 15, hl = lane >> 4;

  // staging role: each thread moves one 16B chunk of the 64x32 B tile
  const int srow = tid >> 2;            // 0..63
  const int schk = tid & 3;             // 0..3 (8 halves each)
  const _Float16* gstage_base = Bt + (size_t)(n0 + srow) * Kdim + schk * 8;
  _Float16* lstage = &btile[srow * BROWPAD + schk * 8];

  float8 accA0 = {}, accA1 = {}, accA2 = {}, accA3 = {};
  float8 accB0 = {}, accB1 = {}, accB2 = {}, accB3 = {};

  for (int kb = 0; kb < Kdim; kb += 32) {
    // ---- stage B strip into LDS ----
    const _Float16* gp = gstage_base + kb;
#if HAVE_ASYNC_LDS
    __builtin_amdgcn_global_load_async_to_lds_b128(
        (as1_i4p)(uintptr_t)gp,
        (as3_i4p)(uint32_t)(uintptr_t)lstage,  // low 32 bits of generic LDS addr = LDS offset
        0, 0);
#else
    *(half8*)lstage = *(const half8*)gp;
#endif
    // ---- A fragments from global (overlaps async copy) ----
    const half16 a0 = load_afrag(A, Kdim, m0, kb, lane);
    const half16 a1 = load_afrag(A, Kdim, m0 + 16, kb, lane);
    if (kb + 32 < Kdim) {  // prefetch next A K-block (global_prefetch_b8)
      __builtin_prefetch(A + (size_t)(m0 + r) * Kdim + kb + 32, 0, 3);
      __builtin_prefetch(A + (size_t)(m0 + 16 + r) * Kdim + kb + 32, 0, 3);
    }
#if HAVE_ASYNC_LDS
    asm volatile("s_wait_asynccnt 0x0" ::: "memory");
#endif
    __syncthreads();
    // ---- 4 shared B-frags, 8 WMMAs ----
    const half16 b0 = *(const half16*)&btile[(0 * 16 + r) * BROWPAD + hl * 16];
    const half16 b1 = *(const half16*)&btile[(1 * 16 + r) * BROWPAD + hl * 16];
    const half16 b2 = *(const half16*)&btile[(2 * 16 + r) * BROWPAD + hl * 16];
    const half16 b3 = *(const half16*)&btile[(3 * 16 + r) * BROWPAD + hl * 16];
    accA0 = wmma_f16(a0, b0, accA0);
    accB0 = wmma_f16(a1, b0, accB0);
    accA1 = wmma_f16(a0, b1, accA1);
    accB1 = wmma_f16(a1, b1, accB1);
    accA2 = wmma_f16(a0, b2, accA2);
    accB2 = wmma_f16(a1, b2, accB2);
    accA3 = wmma_f16(a0, b3, accA3);
    accB3 = wmma_f16(a1, b3, accB3);
    __syncthreads();   // protect btile before next overwrite
  }

#pragma unroll
  for (int j = 0; j < 8; ++j) {
    const size_t rowA = (size_t)(m0 + j + 8 * hl) * Ndim;
    C[rowA + n0 + r]      = accA0[j];
    C[rowA + n0 + 16 + r] = accA1[j];
    C[rowA + n0 + 32 + r] = accA2[j];
    C[rowA + n0 + 48 + r] = accA3[j];
    const size_t rowB = (size_t)(m0 + 16 + j + 8 * hl) * Ndim;
    C[rowB + n0 + r]      = accB0[j];
    C[rowB + n0 + 16 + r] = accB1[j];
    C[rowB + n0 + 32 + r] = accB2[j];
    C[rowB + n0 + 48 + r] = accB3[j];
  }
}

// ---------------- RoPE + relayout [B,S,nh,HD] f32 -> [B,nh,S,HD] f16 ----------------
__global__ void rope_relayout(const float* __restrict__ src, const int* __restrict__ pos,
                              _Float16* __restrict__ dst, int nh) {
  const int idx = blockIdx.x * blockDim.x + threadIdx.x;
  if (idx >= B_ * S_ * nh * 32) return;
  const int d = idx & 31;
  int t = idx >> 5;
  const int h = t % nh; t /= nh;
  const int s = t % S_;
  const int b = t / S_;
  const float p = (float)pos[b * S_ + s];
  const float ang = p * __expf(-(float)d * 0.28782313662425572f);  // ln(10000)/32
  const float c = __cosf(ang), sn = __sinf(ang);
  const size_t srow = (size_t)(b * S_ + s) * (nh * HD_) + h * HD_;
  const float x1 = src[srow + d];
  const float x2 = src[srow + d + 32];
  const size_t drow = ((size_t)(b * nh + h) * S_ + s) * HD_;
  dst[drow + d]      = (_Float16)(x1 * c - x2 * sn);
  dst[drow + d + 32] = (_Float16)(x2 * c + x1 * sn);
}

// ---------------- V relayout [B,S,NKV,HD] f32 -> [B,NKV,HD,S] f16 ----------------
__global__ void v_relayout(const float* __restrict__ src, _Float16* __restrict__ dst) {
  const int idx = blockIdx.x * blockDim.x + threadIdx.x;
  if (idx >= B_ * S_ * NKV_ * HD_) return;
  const int d = idx % HD_;
  int t = idx / HD_;
  const int h = t % NKV_; t /= NKV_;
  const int s = t % S_;
  const int b = t / S_;
  dst[((size_t)(b * NKV_ + h) * HD_ + d) * S_ + s] =
      (_Float16)src[(size_t)(b * S_ + s) * DKV_ + h * HD_ + d];
}

// ---------------- flash attention: one wave per (b, head, 16-query tile) ----------------
__global__ __launch_bounds__(128) void attn_kernel(const _Float16* __restrict__ qh,
                                                   const _Float16* __restrict__ kh,
                                                   const _Float16* __restrict__ vT,
                                                   _Float16* __restrict__ attnh) {
  __shared__ __align__(64) _Float16 pshare[4][16 * 32];
  const int lane = threadIdx.x & 31;
  const int wib  = threadIdx.x >> 5;
  const int wave = (blockIdx.x * blockDim.x + threadIdx.x) >> 5;
  const int QT = S_ / 16;
  const int qt = wave % QT;
  const int h  = (wave / QT) % NH_;
  const int b  = wave / (QT * NH_);
  const int kvh = h / (NH_ / NKV_);
  const _Float16* Qp = qh + (size_t)(b * NH_ + h) * S_ * HD_;
  const _Float16* Kp = kh + (size_t)(b * NKV_ + kvh) * S_ * HD_;
  const _Float16* Vp = vT + (size_t)(b * NKV_ + kvh) * HD_ * S_;
  _Float16* pbuf = pshare[wib];
  const int r = lane & 15, hl = lane >> 4;
  const int q0 = qt * 16;

  const half16 qa0 = load_afrag(Qp, HD_, q0, 0, lane);
  const half16 qa1 = load_afrag(Qp, HD_, q0, 32, lane);

  float8 oacc[4] = {{}, {}, {}, {}};
  float mrun[8], lrun[8];
#pragma unroll
  for (int j = 0; j < 8; ++j) { mrun[j] = -1e30f; lrun[j] = 0.0f; }

  const int kend = q0 + 16;  // causal: keys <= q0+15
  for (int kb = 0; kb < kend; kb += 32) {
    float8 s0 = {}, s1 = {};
    s0 = wmma_f16(qa0, load_bfrag(Kp, HD_, kb,      0,  lane), s0);
    s0 = wmma_f16(qa1, load_bfrag(Kp, HD_, kb,      32, lane), s0);
    s1 = wmma_f16(qa0, load_bfrag(Kp, HD_, kb + 16, 0,  lane), s1);
    s1 = wmma_f16(qa1, load_bfrag(Kp, HD_, kb + 16, 32, lane), s1);
#pragma unroll
    for (int j = 0; j < 8; ++j) {
      const int row = q0 + j + 8 * hl;
      float a0 = (kb + r      <= row) ? s0[j] * 0.125f : -1e30f;  // 1/sqrt(64)
      float a1 = (kb + 16 + r <= row) ? s1[j] * 0.125f : -1e30f;
      float mx = fmaxf(a0, a1);
      mx = fmaxf(mx, __shfl_xor(mx, 1));
      mx = fmaxf(mx, __shfl_xor(mx, 2));
      mx = fmaxf(mx, __shfl_xor(mx, 4));
      mx = fmaxf(mx, __shfl_xor(mx, 8));
      const float mnew  = fmaxf(mrun[j], mx);
      const float alpha = __expf(mrun[j] - mnew);
      const float p0 = __expf(a0 - mnew);
      const float p1 = __expf(a1 - mnew);
      float ps = p0 + p1;
      ps += __shfl_xor(ps, 1);
      ps += __shfl_xor(ps, 2);
      ps += __shfl_xor(ps, 4);
      ps += __shfl_xor(ps, 8);
      lrun[j] = lrun[j] * alpha + ps;
      mrun[j] = mnew;
#pragma unroll
      for (int t = 0; t < 4; ++t) oacc[t][j] *= alpha;
      pbuf[(j + 8 * hl) * 32 + r]      = (_Float16)p0;
      pbuf[(j + 8 * hl) * 32 + 16 + r] = (_Float16)p1;
    }
    asm volatile("s_wait_dscnt 0" ::: "memory");  // wave-local LDS RAW fence
    const half16 pa = load_afrag(pbuf, 32, 0, 0, lane);
    const int koff = kb + hl * 16;
#pragma unroll
    for (int t = 0; t < 4; ++t) {
      const half16 vb = *(const half16*)(Vp + (size_t)(t * 16 + r) * S_ + koff);
      oacc[t] = wmma_f16(pa, vb, oacc[t]);
    }
  }
#pragma unroll
  for (int t = 0; t < 4; ++t)
#pragma unroll
    for (int j = 0; j < 8; ++j) {
      const float ov = oacc[t][j] / lrun[j];
      attnh[(size_t)(b * S_ + q0 + j + 8 * hl) * DQ_ + h * HD_ + t * 16 + r] = (_Float16)ov;
    }
}

// ---------------- host orchestration ----------------
extern "C" void kernel_launch(void* const* d_in, const int* in_sizes, int n_in,
                              void* d_out, int out_size, void* d_ws, size_t ws_size,
                              hipStream_t stream) {
  (void)in_sizes; (void)n_in; (void)out_size; (void)ws_size;
  const float* x  = (const float*)d_in[0];
  const int*   pos = (const int*)d_in[1];
  const float* Wq = (const float*)d_in[2];
  const float* Wk = (const float*)d_in[3];
  const float* Wv = (const float*)d_in[4];
  const float* Wo = (const float*)d_in[5];
  const float* Aq = (const float*)d_in[6];
  const float* Bq = (const float*)d_in[7];
  const float* Ak = (const float*)d_in[8];
  const float* Bk = (const float*)d_in[9];
  const float* Av = (const float*)d_in[10];
  const float* Bv = (const float*)d_in[11];
  const float* Ao = (const float*)d_in[12];
  const float* Bo = (const float*)d_in[13];
  float* out = (float*)d_out;

  char* ws = (char*)d_ws;
  size_t off = 0;
  auto carve = [&](size_t bytes) -> void* {
    void* p = ws + off;
    off += (bytes + 255) & ~(size_t)255;
    return p;
  };
  _Float16* xh  = (_Float16*)carve((size_t)M_ * H_ * 2);
  _Float16* wqe = (_Float16*)carve((size_t)DQ_ * H_ * 2);
  _Float16* wke = (_Float16*)carve((size_t)DKV_ * H_ * 2);
  _Float16* wve = (_Float16*)carve((size_t)DKV_ * H_ * 2);
  _Float16* woe = (_Float16*)carve((size_t)H_ * DQ_ * 2);
  float*    qf  = (float*)carve((size_t)M_ * DQ_ * 4);
  float*    kf  = (float*)carve((size_t)M_ * DKV_ * 4);
  float*    vf  = (float*)carve((size_t)M_ * DKV_ * 4);
  _Float16* qhp = (_Float16*)carve((size_t)M_ * DQ_ * 2);
  _Float16* khp = (_Float16*)carve((size_t)M_ * DKV_ * 2);
  _Float16* vTp = (_Float16*)carve((size_t)M_ * DKV_ * 2);
  _Float16* attnh = (_Float16*)qf;  // reuse qf: dead after rope_relayout(q)

  // 1) fold LoRA into f16 effective weights ([out][in] for B-fragment loads)
  prep_weight<<<(DQ_ * H_ + 255) / 256, 256, 0, stream>>>(Wq, Aq, Bq, wqe, H_, DQ_);
  prep_weight<<<(DKV_ * H_ + 255) / 256, 256, 0, stream>>>(Wk, Ak, Bk, wke, H_, DKV_);
  prep_weight<<<(DKV_ * H_ + 255) / 256, 256, 0, stream>>>(Wv, Av, Bv, wve, H_, DKV_);
  prep_weight<<<(H_ * DQ_ + 255) / 256, 256, 0, stream>>>(Wo, Ao, Bo, woe, DQ_, H_);

  // 2) x -> f16
  to_half_kernel<<<(M_ * H_ + 255) / 256, 256, 0, stream>>>(x, xh, M_ * H_);

  // 3) QKV projections (block-cooperative WMMA, 256x64 C tile per block)
  auto gblocks = [](int M, int N) { return (M / 256) * (N / 64); };
  gemm_wmma<<<gblocks(M_, DQ_), 256, 0, stream>>>(xh, wqe, qf, M_, DQ_, H_);
  gemm_wmma<<<gblocks(M_, DKV_), 256, 0, stream>>>(xh, wke, kf, M_, DKV_, H_);
  gemm_wmma<<<gblocks(M_, DKV_), 256, 0, stream>>>(xh, wve, vf, M_, DKV_, H_);

  // 4) RoPE + relayouts
  rope_relayout<<<(B_ * S_ * NH_ * 32 + 255) / 256, 256, 0, stream>>>(qf, pos, qhp, NH_);
  rope_relayout<<<(B_ * S_ * NKV_ * 32 + 255) / 256, 256, 0, stream>>>(kf, pos, khp, NKV_);
  v_relayout<<<(B_ * S_ * NKV_ * HD_ + 255) / 256, 256, 0, stream>>>(vf, vTp);

  // 5) causal flash attention (WMMA)
  attn_kernel<<<(B_ * NH_ * (S_ / 16) * 32) / 128, 128, 0, stream>>>(qhp, khp, vTp, attnh);

  // 6) output projection -> fp32 d_out
  gemm_wmma<<<gblocks(M_, H_), 256, 0, stream>>>(attnh, woe, out, M_, H_, DQ_);
}
